// MSP_ResNet_60567628808656
// MI455X (gfx1250) — matrix operand order repaired and checked by reading, not compiled, not run on hardware
//
#include <hip/hip_runtime.h>

typedef _Float16 v2h  __attribute__((ext_vector_type(2)));
typedef _Float16 v8h  __attribute__((ext_vector_type(8)));
typedef _Float16 v16h __attribute__((ext_vector_type(16)));
typedef float    v8f  __attribute__((ext_vector_type(8)));
typedef unsigned int u32x4 __attribute__((ext_vector_type(4)));
typedef int          i32x8 __attribute__((ext_vector_type(8)));
typedef int          i32x4 __attribute__((ext_vector_type(4)));

#define EPSV 1e-5f
#define Cc   512
#define HIDc 128
#define HWc  3136
#define TN   64

#if __has_builtin(__builtin_amdgcn_tensor_load_to_lds)
#define HAVE_TDM 1
#else
#define HAVE_TDM 0
#endif

// ---------------- prep: fold BN scale into W1, convert weights to f16 ----------------
__global__ __launch_bounds__(256) void prep_kernel(
    const float* __restrict__ w1_rgb, const float* __restrict__ rm_rgb, const float* __restrict__ rv_rgb,
    const float* __restrict__ g_rgb,  const float* __restrict__ b_rgb,  const float* __restrict__ w2_rgb,
    const float* __restrict__ w1_inf, const float* __restrict__ rm_inf, const float* __restrict__ rv_inf,
    const float* __restrict__ g_inf,  const float* __restrict__ b_inf,  const float* __restrict__ w2_inf,
    _Float16* __restrict__ w1h_rgb, _Float16* __restrict__ w1h_inf,
    _Float16* __restrict__ w2h_rgb, _Float16* __restrict__ w2h_inf,
    float* __restrict__ sh_rgb, float* __restrict__ sh_inf)
{
    int i = blockIdx.x * 256 + threadIdx.x;      // 0..65535
    int o = i >> 9;                              // W1 row (HID)
    float scr = g_rgb[o] * rsqrtf(rv_rgb[o] + EPSV);
    float sci = g_inf[o] * rsqrtf(rv_inf[o] + EPSV);
    w1h_rgb[i] = (_Float16)(w1_rgb[i] * scr);
    w1h_inf[i] = (_Float16)(w1_inf[i] * sci);
    w2h_rgb[i] = (_Float16)w2_rgb[i];
    w2h_inf[i] = (_Float16)w2_inf[i];
    if ((i & 511) == 0) {
        sh_rgb[o] = b_rgb[o] - rm_rgb[o] * scr;
        sh_inf[o] = b_inf[o] - rm_inf[o] * sci;
    }
}

#if HAVE_TDM
// Issue one TDM 2-D tile load: 64 (pixels, contiguous) x 64 (C rows, stride 3136 f32)
// into LDS at lds_off with a 68-float row pitch via D# padding (64 DWORDs then +4 DWORDs).
__device__ __forceinline__ void tdm_load_chunk(const float* gsrc, unsigned lds_off) {
    unsigned long long ga = (unsigned long long)(uintptr_t)gsrc;
    u32x4 g0;
    g0[0] = 1u;                                   // count=1, user descriptor
    g0[1] = lds_off;                              // lds_addr (bytes)
    g0[2] = (unsigned)(ga & 0xFFFFFFFFu);         // global_addr[31:0]
    g0[3] = (unsigned)(ga >> 32) | (2u << 30);    // global_addr[56:32] | type=2
    i32x8 g1;
    g1[0] = (2 << 16)      // data_size = 4B
          | (1 << 20)      // pad_enable
          | (5 << 22)      // pad_interval: 64 DWORDs
          | (3 << 25);     // pad_amount: 4 DWORDs
    g1[1] = (int)(3136u << 16);   // tensor_dim0 = 3136 (low16 into [63:48])
    g1[2] = (int)(512u << 16);    // tensor_dim0 hi=0 | tensor_dim1 = 512
    g1[3] = (int)(64u << 16);     // tensor_dim1 hi=0 | tile_dim0 = 64
    g1[4] = 64;                   // tile_dim1 = 64, tile_dim2 = 0
    g1[5] = 3136;                 // tensor_dim0_stride[31:0]
    g1[6] = 0;                    // stride0 hi | stride1 lo
    g1[7] = 0;
    i32x4 z4 = {0, 0, 0, 0};
#if defined(__clang_major__) && (__clang_major__ >= 23)
    i32x8 z8 = {0, 0, 0, 0, 0, 0, 0, 0};
    __builtin_amdgcn_tensor_load_to_lds(g0, g1, z4, z4, z8, 0);
#else
    __builtin_amdgcn_tensor_load_to_lds(g0, g1, z4, z4, 0);
#endif
}
#endif

// ---------------- fused: conv1x1 -> BN(fold)+ReLU -> conv1x1, branch-selected ----------------
__global__ __launch_bounds__(256) void fused_kernel(
    const float* __restrict__ x, const int* __restrict__ mod,
    const _Float16* __restrict__ w1h_rgb, const _Float16* __restrict__ w1h_inf,
    const _Float16* __restrict__ w2h_rgb, const _Float16* __restrict__ w2h_inf,
    const float* __restrict__ sh_rgb, const float* __restrict__ sh_inf,
    float* __restrict__ out)
{
    __shared__ float    xs[2][64 * 68];  // double-buffered f32 stage chunk [k'][n], pitch 68
    __shared__ _Float16 xh[64 * 72];     // f16 [n][k'] chunk, pitch 72 (B-operand layout)
    __shared__ _Float16 hh[64 * 136];    // f16 H [n][k], pitch 136

    const int b    = blockIdx.y;
    const int px0  = blockIdx.x * TN;
    const int tid  = threadIdx.x;
    const int lane = tid & 31;
    const int w    = tid >> 5;       // wave 0..7
    const int colL = lane & 15;      // column / row-within-16
    const int hiL  = lane >> 4;      // lane-half select

    const bool rgb = (mod[b] == 1);
    const _Float16* w1p = rgb ? w1h_rgb : w1h_inf;
    const _Float16* w2p = rgb ? w2h_rgb : w2h_inf;
    const float*    shp = rgb ? sh_rgb : sh_inf;

    const float* xb = x + ((size_t)b * Cc) * HWc + px0;

    const v8f zacc = {0.f,0.f,0.f,0.f,0.f,0.f,0.f,0.f};
    v8f acc1[4];
    #pragma unroll
    for (int nb = 0; nb < 4; ++nb) acc1[nb] = zacc;

#if HAVE_TDM
    // prologue: async-fetch chunk 0 into buffer 0
    if (w == 0) tdm_load_chunk(xb, (unsigned)(uintptr_t)&xs[0][0]);
#endif

    // ---------------- GEMM1 over K=512 in 8 chunks of 64 ----------------
    for (int kc = 0; kc < 8; ++kc) {
        const int buf = kc & 1;
#if HAVE_TDM
        if (w == 0) __builtin_amdgcn_s_wait_tensorcnt(0);   // chunk kc landed in xs[buf]
        __syncthreads();
        // overlap: start DMA of chunk kc+1 into the other buffer while we transpose+MMA
        if (w == 0 && kc < 7)
            tdm_load_chunk(xb + (size_t)(kc + 1) * 64 * HWc,
                           (unsigned)(uintptr_t)&xs[buf ^ 1][0]);
#else
        // manual staging fallback: coalesced float4 loads
        {
            int r0 = tid >> 4;
            int p0 = (tid & 15) * 4;
            #pragma unroll
            for (int s = 0; s < 4; ++s) {
                int r = r0 + 16 * s;
                float4 v = *(const float4*)(xb + (size_t)(kc * 64 + r) * HWc + p0);
                *(float4*)&xs[buf][r * 68 + p0] = v;
            }
        }
        __syncthreads();
#endif
        // transpose + convert to f16 [n][k'] (one b128 LDS store per 8 k-values)
        {
            int n  = tid & 63;
            int ks = (tid >> 6) * 8;
            #pragma unroll
            for (int i2 = 0; i2 < 2; ++i2) {
                int k0 = ks + 32 * i2;
                v8h hv;
                #pragma unroll
                for (int j = 0; j < 8; ++j) hv[j] = (_Float16)xs[buf][(k0 + j) * 68 + n];
                *(v8h*)&xh[n * 72 + k0] = hv;
            }
        }
        __syncthreads();
        // 2 k-steps of 32 on this chunk
        #pragma unroll
        for (int kk = 0; kk < 2; ++kk) {
            // A fragment: W1 rows 16w..16w+15, ISA 16-bit A layout (K 0-7/16-23 | 8-15/24-31)
            const _Float16* arow = w1p + (size_t)(16 * w + colL) * Cc + kc * 64 + kk * 32 + hiL * 8;
            v8h a0 = *(const v8h*)arow;
            v8h a1 = *(const v8h*)(arow + 16);
            v16h A = __builtin_shufflevector(a0, a1, 0,1,2,3,4,5,6,7,8,9,10,11,12,13,14,15);
            #pragma unroll
            for (int nb = 0; nb < 4; ++nb) {
                // B fragment: same layout as A (K 0-7/16-23 | 8-15/24-31)
                const _Float16* bp = &xh[(16 * nb + colL) * 72 + kk * 32 + hiL * 8];
                v8h b0 = *(const v8h*)bp;
                v8h b1 = *(const v8h*)(bp + 16);
                v16h B = __builtin_shufflevector(b0, b1, 0,1,2,3,4,5,6,7,8,9,10,11,12,13,14,15);
                acc1[nb] = __builtin_amdgcn_wmma_f32_16x16x32_f16(
                    false, A, false, B, (short)0, acc1[nb], false, false);
            }
        }
    }

    // ---------------- epilogue: +shift, ReLU, f16 -> hh ----------------
    {
        float sh[8];
        #pragma unroll
        for (int j = 0; j < 8; ++j) sh[j] = shp[16 * w + hiL * 8 + j];
        #pragma unroll
        for (int nb = 0; nb < 4; ++nb) {
            int n = 16 * nb + colL;
            #pragma unroll
            for (int j = 0; j < 8; j += 2) {
                float v0 = fmaxf(acc1[nb][j]     + sh[j],     0.f);
                float v1 = fmaxf(acc1[nb][j + 1] + sh[j + 1], 0.f);
                v2h p; p[0] = (_Float16)v0; p[1] = (_Float16)v1;
                *(v2h*)&hh[n * 136 + 16 * w + hiL * 8 + j] = p;
            }
        }
    }
    __syncthreads();

    // ---------------- GEMM2: OUT(512x64) = W2(512x128) x H(128x64) ----------------
    v8f acc2[4][4];
    #pragma unroll
    for (int mb = 0; mb < 4; ++mb)
        #pragma unroll
        for (int nb = 0; nb < 4; ++nb) acc2[mb][nb] = zacc;

    #pragma unroll
    for (int k = 0; k < 4; ++k) {
        v16h Af[4];
        #pragma unroll
        for (int mb = 0; mb < 4; ++mb) {
            const _Float16* arow = w2p + (size_t)(64 * w + 16 * mb + colL) * HIDc + k * 32 + hiL * 8;
            v8h a0 = *(const v8h*)arow;
            v8h a1 = *(const v8h*)(arow + 16);
            Af[mb] = __builtin_shufflevector(a0, a1, 0,1,2,3,4,5,6,7,8,9,10,11,12,13,14,15);
        }
        #pragma unroll
        for (int nb = 0; nb < 4; ++nb) {
            const _Float16* bp = &hh[(16 * nb + colL) * 136 + k * 32 + hiL * 8];
            v8h b0 = *(const v8h*)bp;
            v8h b1 = *(const v8h*)(bp + 16);
            v16h B = __builtin_shufflevector(b0, b1, 0,1,2,3,4,5,6,7,8,9,10,11,12,13,14,15);
            #pragma unroll
            for (int mb = 0; mb < 4; ++mb)
                acc2[mb][nb] = __builtin_amdgcn_wmma_f32_16x16x32_f16(
                    false, Af[mb], false, B, (short)0, acc2[mb][nb], false, false);
        }
    }

    // store fp32 output
    float* ob = out + ((size_t)b * Cc) * HWc + px0;
    #pragma unroll
    for (int mb = 0; mb < 4; ++mb)
        #pragma unroll
        for (int nb = 0; nb < 4; ++nb)
            #pragma unroll
            for (int j = 0; j < 8; ++j) {
                int row = 64 * w + 16 * mb + hiL * 8 + j;
                int col = 16 * nb + colL;
                ob[(size_t)row * HWc + col] = acc2[mb][nb][j];
            }
}

extern "C" void kernel_launch(void* const* d_in, const int* in_sizes, int n_in,
                              void* d_out, int out_size, void* d_ws, size_t ws_size,
                              hipStream_t stream) {
    (void)in_sizes; (void)n_in; (void)out_size; (void)ws_size;
    const float* x       = (const float*)d_in[0];
    const int*   mod     = (const int*)  d_in[1];
    const float* w1_rgb  = (const float*)d_in[2];
    const float* rm_rgb  = (const float*)d_in[3];
    const float* rv_rgb  = (const float*)d_in[4];
    const float* g_rgb   = (const float*)d_in[5];
    const float* b_rgb   = (const float*)d_in[6];
    const float* w2_rgb  = (const float*)d_in[7];
    const float* w1_inf  = (const float*)d_in[8];
    const float* rm_inf  = (const float*)d_in[9];
    const float* rv_inf  = (const float*)d_in[10];
    const float* g_inf   = (const float*)d_in[11];
    const float* b_inf   = (const float*)d_in[12];
    const float* w2_inf  = (const float*)d_in[13];

    char* ws = (char*)d_ws;
    _Float16* w1h_rgb = (_Float16*)(ws);
    _Float16* w1h_inf = (_Float16*)(ws + 131072);
    _Float16* w2h_rgb = (_Float16*)(ws + 262144);
    _Float16* w2h_inf = (_Float16*)(ws + 393216);
    float*    sh_rgb  = (float*)   (ws + 524288);
    float*    sh_inf  = (float*)   (ws + 524800);

    prep_kernel<<<256, 256, 0, stream>>>(
        w1_rgb, rm_rgb, rv_rgb, g_rgb, b_rgb, w2_rgb,
        w1_inf, rm_inf, rv_inf, g_inf, b_inf, w2_inf,
        w1h_rgb, w1h_inf, w2h_rgb, w2h_inf, sh_rgb, sh_inf);

    dim3 grid(HWc / TN, 32);   // 49 x 32
    fused_kernel<<<grid, 256, 0, stream>>>(
        x, mod, w1h_rgb, w1h_inf, w2h_rgb, w2h_inf, sh_rgb, sh_inf, (float*)d_out);
}
